// ProbAttention_14594298872559
// MI455X (gfx1250) — matrix operand order repaired and checked by reading, not compiled, not run on hardware
//
#include <hip/hip_runtime.h>
#include <hip/hip_bf16.h>

typedef __attribute__((ext_vector_type(16))) _Float16 v16h;
typedef __attribute__((ext_vector_type(8)))  float    v8f;

#define B_ 4
#define L_ 2048
#define H_ 8
#define D_ 64
#define N_ 1024   // L_ * SAMPLE_RATIO

__device__ __forceinline__ v8f wmma_f16(v16h a, v16h b, v8f c) {
  // D = A(16x32 f16) * B(32x16 f16) + C(16x16 f32)
  return __builtin_amdgcn_wmma_f32_16x16x32_f16(
      /*neg_a=*/false, a, /*neg_b=*/false, b,
      /*c_mod=*/(short)0, c, /*reuse_a=*/false, /*reuse_b=*/false);
}

// ---------------------------------------------------------------- kernel 0
__global__ void zero_imp_kernel(float* imp) {
  int i = blockIdx.x * blockDim.x + threadIdx.x;
  if (i < L_) imp[i] = 0.0f;
}

// ---------------------------------------------------------------- kernel 1
// importance[k] += sum_{b,h,q} |Q[b,q,h,:] . K[b,k,h,:]|
// grid: (L/256, B*H), block 128 (4 waves); wave owns 64 key columns held
// resident in registers (8 x v16h), so each Q-tile load feeds 8 WMMAs.
__global__ __launch_bounds__(128)
void importance_kernel(const float* __restrict__ Q, const float* __restrict__ K,
                       float* __restrict__ imp) {
  const int bh   = blockIdx.y;
  const int b    = bh / H_, h = bh % H_;
  const int wave = threadIdx.x >> 5;
  const int lane = threadIdx.x & 31;
  const int m    = lane & 15;       // row (A) / column (B/C) within tile
  const int hs   = lane >> 4;       // contraction-half select
  const int kbase = blockIdx.x * 256 + wave * 64;

  // B operands: 4 column groups x (d0-31, d32-63)
  v16h kb0[4], kb1[4];
  #pragma unroll
  for (int g = 0; g < 4; ++g) {
    const float* krow = K + (((size_t)b * L_ + (kbase + g * 16 + m)) * H_ + h) * D_;
    #pragma unroll
    for (int i = 0; i < 16; ++i) {
      kb0[g][i] = (_Float16)krow[hs * 16 + i];
      kb1[g][i] = (_Float16)krow[32 + hs * 16 + i];
    }
  }

  float colsum[4] = {0.0f, 0.0f, 0.0f, 0.0f};
  for (int qt = 0; qt < L_ / 16; ++qt) {
    const float* qrow = Q + (((size_t)b * L_ + (qt * 16 + m)) * H_ + h) * D_;
    v16h qa0, qa1;
    #pragma unroll
    for (int i = 0; i < 8; ++i) {
      qa0[i]     = (_Float16)qrow[hs * 16 + i];
      qa0[8 + i] = (_Float16)qrow[hs * 16 + 8 + i];
      qa1[i]     = (_Float16)qrow[32 + hs * 16 + i];
      qa1[8 + i] = (_Float16)qrow[32 + hs * 16 + 8 + i];
    }
    #pragma unroll
    for (int g = 0; g < 4; ++g) {
      v8f c = {};
      c = wmma_f16(qa0, kb0[g], c);
      c = wmma_f16(qa1, kb1[g], c);
      float s = 0.0f;
      #pragma unroll
      for (int j = 0; j < 8; ++j) s += fabsf(c[j]);
      colsum[g] += s;
    }
  }
  // lanes l and l+16 hold the same column (rows 0-7 vs 8-15)
  #pragma unroll
  for (int g = 0; g < 4; ++g) {
    float s = colsum[g] + __shfl_xor(colsum[g], 16, 32);
    if (lane < 16) atomicAdd(&imp[kbase + g * 16 + lane], s);
  }
}

// ---------------------------------------------------------------- kernel 2
// Bitonic sort of 2048 (value,index) pairs descending; emit first 1024 idx.
__global__ __launch_bounds__(1024)
void topk_kernel(const float* __restrict__ imp, int* __restrict__ idx) {
  __shared__ float sv[L_];
  __shared__ int   si[L_];
  const int t = threadIdx.x;
  for (int i = t; i < L_; i += 1024) { sv[i] = imp[i]; si[i] = i; }
  __syncthreads();
  for (int k = 2; k <= L_; k <<= 1) {
    for (int j = k >> 1; j > 0; j >>= 1) {
      int i = ((t / j) * (2 * j)) + (t % j);
      int p = i + j;
      bool desc = ((i & k) == 0);
      float a = sv[i], c = sv[p];
      int   ia = si[i], ic = si[p];
      bool aFirst = (a > c) || (a == c && ia < ic);  // descending order
      if (desc ? !aFirst : aFirst) {
        sv[i] = c; sv[p] = a; si[i] = ic; si[p] = ia;
      }
      __syncthreads();
    }
  }
  for (int i = t; i < N_; i += 1024) idx[i] = si[i];
}

// ---------------------------------------------------------------- kernel 3
// Sampled attention, flash style. grid: (N/128, B*H), block 128 (4 waves);
// each wave owns 32 q rows (2 A-tiles), so every LDS B operand feeds 2 WMMAs.
__global__ __launch_bounds__(128)
void attn_kernel(const float* __restrict__ Q, const float* __restrict__ K,
                 const float* __restrict__ V, const int* __restrict__ idx,
                 float* __restrict__ out) {
  const float scale = 0.125f;  // 1/sqrt(64)
  const int bh   = blockIdx.y;
  const int b    = bh / H_, h = bh % H_;
  const int wave = threadIdx.x >> 5;
  const int lane = threadIdx.x & 31;
  const int m    = lane & 15;
  const int hs   = lane >> 4;
  const int qbase = blockIdx.x * 128 + wave * 32;

  __shared__ _Float16 sK[32][72];        // gathered K block, row-major, padded
  __shared__ _Float16 sVt[64][40];       // gathered V block, transposed [d][k]
  __shared__ _Float16 sP[4][32][32];     // per-wave P relayout scratch (2 tiles)

  // A operands for this wave's 2x16 query rows (gathered, pre-scaled)
  v16h qa0[2], qa1[2];
  #pragma unroll
  for (int u = 0; u < 2; ++u) {
    const int qsel = idx[qbase + u * 16 + m];
    const float* qp = Q + (((size_t)b * L_ + qsel) * H_ + h) * D_;
    #pragma unroll
    for (int i = 0; i < 8; ++i) {
      qa0[u][i]     = (_Float16)(qp[hs * 16 + i] * scale);
      qa0[u][8 + i] = (_Float16)(qp[hs * 16 + 8 + i] * scale);
      qa1[u][i]     = (_Float16)(qp[32 + hs * 16 + i] * scale);
      qa1[u][8 + i] = (_Float16)(qp[32 + hs * 16 + 8 + i] * scale);
    }
  }

  float mrow[2][8], lrow[2][8];
  v8f acc[2][4] = {};
  #pragma unroll
  for (int u = 0; u < 2; ++u)
    #pragma unroll
    for (int j = 0; j < 8; ++j) { mrow[u][j] = -1e30f; lrow[u][j] = 0.0f; }

  for (int kb = 0; kb < N_; kb += 32) {
    __syncthreads();
    // cooperative gather of K/V block: 32 rows x 64 d, f32 -> f16
    for (int e = threadIdx.x; e < 32 * 64; e += blockDim.x) {
      int r = e >> 6, c = e & 63;
      int grow = idx[kb + r];
      const float* kp = K + (((size_t)b * L_ + grow) * H_ + h) * D_;
      const float* vp = V + (((size_t)b * L_ + grow) * H_ + h) * D_;
      sK[r][c]  = (_Float16)kp[c];
      sVt[c][r] = (_Float16)vp[c];
    }
    __syncthreads();

    // S tiles (16q x 16k), contraction over d=64; B ops shared by both A tiles
    v16h b00, b01, b10, b11;
    #pragma unroll
    for (int i = 0; i < 16; ++i) {
      b00[i] = sK[m][hs * 16 + i];
      b01[i] = sK[m][32 + hs * 16 + i];
      b10[i] = sK[16 + m][hs * 16 + i];
      b11[i] = sK[16 + m][32 + hs * 16 + i];
    }
    v8f s0[2], s1[2];
    #pragma unroll
    for (int u = 0; u < 2; ++u) {
      v8f t0 = {}, t1 = {};
      t0 = wmma_f16(qa0[u], b00, t0);
      t0 = wmma_f16(qa1[u], b01, t0);
      t1 = wmma_f16(qa0[u], b10, t1);
      t1 = wmma_f16(qa1[u], b11, t1);
      s0[u] = t0; s1[u] = t1;
    }

    // online softmax per A-tile; row j+8*hs stats in slot j of each lane
    #pragma unroll
    for (int u = 0; u < 2; ++u) {
      #pragma unroll
      for (int j = 0; j < 8; ++j) {
        float v = fmaxf(s0[u][j], s1[u][j]);
        #pragma unroll
        for (int off = 1; off < 16; off <<= 1) v = fmaxf(v, __shfl_xor(v, off, 32));
        float nm = fmaxf(mrow[u][j], v);
        float corr = __expf(mrow[u][j] - nm);
        lrow[u][j] *= corr;
        #pragma unroll
        for (int t = 0; t < 4; ++t) acc[u][t][j] *= corr;
        mrow[u][j] = nm;
        float p0 = __expf(s0[u][j] - nm);
        float p1 = __expf(s1[u][j] - nm);
        float ps = p0 + p1;
        #pragma unroll
        for (int off = 1; off < 16; off <<= 1) ps += __shfl_xor(ps, off, 32);
        lrow[u][j] += ps;
        s0[u][j] = p0; s1[u][j] = p1;
      }
    }

    // relayout P: C-layout -> A-layout via wave-private LDS
    #pragma unroll
    for (int u = 0; u < 2; ++u)
      #pragma unroll
      for (int j = 0; j < 8; ++j) {
        int r = u * 16 + j + 8 * hs;
        sP[wave][r][m]      = (_Float16)s0[u][j];
        sP[wave][r][16 + m] = (_Float16)s1[u][j];
      }
    asm volatile("s_wait_dscnt 0" ::: "memory");  // wave-local LDS RAW fence
    v16h pa[2];
    #pragma unroll
    for (int u = 0; u < 2; ++u)
      #pragma unroll
      for (int i = 0; i < 8; ++i) {
        pa[u][i]     = sP[wave][u * 16 + m][hs * 16 + i];
        pa[u][8 + i] = sP[wave][u * 16 + m][hs * 16 + 8 + i];
      }

    // P(16x32) . V(32x64): 4 d-tiles, each B operand feeds both A tiles
    #pragma unroll
    for (int t = 0; t < 4; ++t) {
      v16h vb;
      #pragma unroll
      for (int i = 0; i < 16; ++i) vb[i] = sVt[16 * t + m][hs * 16 + i];
      acc[0][t] = wmma_f16(pa[0], vb, acc[0][t]);
      acc[1][t] = wmma_f16(pa[1], vb, acc[1][t]);
    }
  }

  // epilogue: divide by row sums, scatter to (B, N, H, D)
  #pragma unroll
  for (int u = 0; u < 2; ++u) {
    float inv[8];
    #pragma unroll
    for (int j = 0; j < 8; ++j) inv[j] = 1.0f / lrow[u][j];
    #pragma unroll
    for (int t = 0; t < 4; ++t) {
      #pragma unroll
      for (int j = 0; j < 8; ++j) {
        int r = u * 16 + j + 8 * hs;
        int d = 16 * t + m;
        out[(((size_t)b * N_ + (qbase + r)) * H_ + h) * D_ + d] = acc[u][t][j] * inv[j];
      }
    }
  }
}

// ---------------------------------------------------------------- launch
extern "C" void kernel_launch(void* const* d_in, const int* in_sizes, int n_in,
                              void* d_out, int out_size, void* d_ws, size_t ws_size,
                              hipStream_t stream) {
  const float* Q = (const float*)d_in[0];
  const float* K = (const float*)d_in[1];
  const float* V = (const float*)d_in[2];
  float* out = (float*)d_out;

  float* imp = (float*)d_ws;                               // 2048 f32
  int*   idx = (int*)((char*)d_ws + L_ * sizeof(float));   // 1024 i32

  zero_imp_kernel<<<dim3((L_ + 255) / 256), 256, 0, stream>>>(imp);
  importance_kernel<<<dim3(L_ / 256, B_ * H_), 128, 0, stream>>>(Q, K, imp);
  topk_kernel<<<dim3(1), 1024, 0, stream>>>(imp, idx);
  attn_kernel<<<dim3(N_ / 128, B_ * H_), 128, 0, stream>>>(Q, K, V, idx, out);
}
